// TreeLSTM_2ary_61460982005924
// MI455X (gfx1250) — compile-verified, matrix operand order checked by reading, not run on hardware
//
#include <hip/hip_runtime.h>
#include <hip/hip_bf16.h>
#include <stdint.h>

// ---------------- problem constants ----------------
#define HID     256
#define NCLASS  5
#define DEPTH   11
#define BATCH   128
#define N_NODES ((1 << DEPTH) - 1)   // 2047

// Packed big-GEMM geometry: A row = [x(256) | hL(256) | hR(256)] -> K = 768
// Output cols = 5 gates * 256 = 1280 (gate order: i, o, u, f0, f1; W_f duplicated for f0/f1)
#define KDIM     768
#define NDIM     1280
#define NT_TILES (NDIM / 16)   // 80
#define KT_TILES (KDIM / 32)   // 24

#define ROWS_WG  32            // node-instances per workgroup (2 WMMA row-tiles)
#define THREADS  512           // 16 waves; wave w owns n-tiles {g*16 + w}

typedef __attribute__((ext_vector_type(16))) __bf16 bf16x16;
typedef __attribute__((ext_vector_type(8)))  float  floatx8;

union FragB16 {
  bf16x16        v;
  uint32_t       u32[8];
  uint4          u4[2];
  unsigned short u16[16];
};

__device__ __forceinline__ unsigned short f32_to_bf16(float f) {
  union { float f; uint32_t u; } cv; cv.f = f;
  uint32_t u = cv.u;
  u += 0x7FFFu + ((u >> 16) & 1u);          // round-to-nearest-even
  return (unsigned short)(u >> 16);
}

// =====================================================================
// Kernel 1: pack + swizzle all 14 weight matrices (f32) into a single
// bf16 [1280 x 768] matrix stored directly in per-lane WMMA-B fragment
// order: flat index = (((nt*24 + kt)*32 + lane)*16 + e).
//   col = nt*16 + (lane&15); kb = lane<16 ? 0 : 8
//   k(e) = kt*32 + (e<8 ? kb+e : 16+kb+(e-8))
// =====================================================================
__global__ __launch_bounds__(256)
void prep_weights(const float* __restrict__ W_i, const float* __restrict__ U_i,
                  const float* __restrict__ W_f, const float* __restrict__ U_f,
                  const float* __restrict__ W_o, const float* __restrict__ U_o,
                  const float* __restrict__ W_u, const float* __restrict__ U_u,
                  unsigned short* __restrict__ Wswz) {
  const int t = blockIdx.x * blockDim.x + threadIdx.x;   // one thread = one 32B fragment lane
  const int TOT = NT_TILES * KT_TILES * 32;              // 61440
  if (t >= TOT) return;
  const int lane = t & 31;
  const int kt   = (t >> 5) % KT_TILES;
  const int nt   = t / (KT_TILES * 32);
  const int col  = lane & 15;
  const int kb   = (lane < 16) ? 0 : 8;
  const int n    = nt * 16 + col;
  const int g    = n >> 8;     // gate: 0=i 1=o 2=u 3=f0 4=f1
  const int j    = n & 255;    // output hidden index

  unsigned short outv[16];
#pragma unroll
  for (int e = 0; e < 16; ++e) {
    const int k = kt * 32 + ((e < 8) ? (kb + e) : (16 + kb + (e - 8)));
    float v;
    if (k < 256) {                        // W * x  part
      const int kk = k;
      const float* W = (g == 0) ? W_i : (g == 1) ? W_o : (g == 2) ? W_u : W_f;
      v = W[j * 256 + kk];
    } else if (k < 512) {                 // U * hL part
      const int kk = k - 256;
      switch (g) {
        case 0:  v = U_i[(0 * 256 + j) * 256 + kk]; break;
        case 1:  v = U_o[(0 * 256 + j) * 256 + kk]; break;
        case 2:  v = U_u[(0 * 256 + j) * 256 + kk]; break;
        case 3:  v = U_f[((0 * 2 + 0) * 256 + j) * 256 + kk]; break;   // U_f[0][0]
        default: v = U_f[((1 * 2 + 0) * 256 + j) * 256 + kk]; break;   // U_f[1][0]
      }
    } else {                              // U * hR part
      const int kk = k - 512;
      switch (g) {
        case 0:  v = U_i[(1 * 256 + j) * 256 + kk]; break;
        case 1:  v = U_o[(1 * 256 + j) * 256 + kk]; break;
        case 2:  v = U_u[(1 * 256 + j) * 256 + kk]; break;
        case 3:  v = U_f[((0 * 2 + 1) * 256 + j) * 256 + kk]; break;   // U_f[0][1]
        default: v = U_f[((1 * 2 + 1) * 256 + j) * 256 + kk]; break;   // U_f[1][1]
      }
    }
    outv[e] = f32_to_bf16(v);
  }
#pragma unroll
  for (int e = 0; e < 16; ++e) Wswz[(size_t)t * 16 + e] = outv[e];
}

// =====================================================================
// Kernel 2: one tree level. 32 rows (node-instances) per workgroup,
// 512 threads = 16 waves. Wave w owns n-tiles {g*16 + w}, g = 0..4:
// after the K loop every lane holds all five gate pre-activations for
// the SAME (row, j)  ->  pointwise is entirely wave-local.
// B fragments are double-buffered (bcur/bnxt) and each feeds 2 WMMAs
// (two row-tiles), so loads overlap matrix ops instead of draining.
// =====================================================================
#define A_STRIDE 776   // ushorts per A-row in LDS (768 + 8 pad -> conflict-free)
#define H_STRIDE 260   // floats per h-row in LDS (aliased over A region)

__global__ __launch_bounds__(THREADS)
void level_kernel(const float* __restrict__ x,
                  const unsigned short* __restrict__ Wswz,
                  const float* __restrict__ b_i, const float* __restrict__ b_f,
                  const float* __restrict__ b_o, const float* __restrict__ b_u,
                  const float* __restrict__ W_s, const float* __restrict__ b_s,
                  unsigned short* __restrict__ h_ws, float* __restrict__ c_ws,
                  float* __restrict__ out, int lvl) {
  const int  M    = 1 << lvl;
  const int  s    = M - 1;
  const int  e    = 2 * M - 1;
  const bool leaf = (lvl == DEPTH - 1);
  const int  tid  = threadIdx.x;
  const int  lane = tid & 31;
  const int  w    = tid >> 5;            // wave 0..15
  const int  r0   = blockIdx.x * ROWS_WG;

  // A-staging region (bf16) aliased with classifier h region (f32):
  __shared__ __align__(16) char smem[ROWS_WG * A_STRIDE * 2];   // 49664 B
  unsigned short* A_sh = (unsigned short*)smem;
  float*          h_sh = (float*)smem;
  __shared__ float logit_sh[ROWS_WG * NCLASS];

  // ---- stage A tile: x (cvt f32->bf16) and children h (bf16 copy) ----
  {
    const int colt = tid & 255;          // column 0..255
    const int rp   = tid >> 8;           // row parity 0/1
    for (int rr = 0; rr < ROWS_WG / 2; ++rr) {
      const int row  = rr * 2 + rp;
      const int r    = r0 + row;
      const int b    = r >> lvl;
      const int m    = r & (M - 1);
      const int node = s + m;
      const size_t xbase = ((size_t)b * N_NODES + node) * HID;
      A_sh[row * A_STRIDE + colt] = f32_to_bf16(x[xbase + colt]);
      if (!leaf) {
        const int cl = e + 2 * m;
        const size_t hl = ((size_t)b * N_NODES + cl) * HID;
        const size_t hr = ((size_t)b * N_NODES + cl + 1) * HID;
        A_sh[row * A_STRIDE + 256 + colt] = h_ws[hl + colt];
        A_sh[row * A_STRIDE + 512 + colt] = h_ws[hr + colt];
      }
    }
  }

  const int KT   = leaf ? (256 / 32) : KT_TILES;  // leaf: hL=hR=0 -> K=256
  const int arow = lane & 15;
  const int kb   = (lane < 16) ? 0 : 8;

  // B fragment address helper: nt = g*16 + w
  // (kt == KT is a harmless in-bounds pipeline overrun: data is loaded but never used)
  #define BOFF(g_, kt_) ((((size_t)((g_) * 16 + w)) * KT_TILES + (size_t)(kt_)) * 32 + lane) * 16

  // prime the B pipeline before the barrier (independent of LDS)
  FragB16 bcur;
  bcur.u4[0] = *(const uint4*)(Wswz + BOFF(0, 0));
  bcur.u4[1] = *(const uint4*)(Wswz + BOFF(0, 0) + 8);

  __syncthreads();

  // ---- GEMM: [32 x K] x [K x 1280] with v_wmma_f32_16x16x32_bf16 ----
  floatx8 acc[10];   // [g*2 + row-tile]
#pragma unroll
  for (int i = 0; i < 10; ++i)
#pragma unroll
    for (int p = 0; p < 8; ++p) acc[i][p] = 0.0f;

  for (int kt = 0; kt < KT; ++kt) {
    // A fragments for both row-tiles from LDS (dword reads, 4-byte aligned)
    FragB16 af0, af1;
    {
      const unsigned short* ab0 = &A_sh[(arow)      * A_STRIDE + kt * 32 + kb];
      const unsigned short* ab1 = &A_sh[(16 + arow) * A_STRIDE + kt * 32 + kb];
#pragma unroll
      for (int vv = 0; vv < 4; ++vv) {
        af0.u32[vv]     = *(const uint32_t*)(ab0 + 2 * vv);
        af0.u32[4 + vv] = *(const uint32_t*)(ab0 + 16 + 2 * vv);
        af1.u32[vv]     = *(const uint32_t*)(ab1 + 2 * vv);
        af1.u32[4 + vv] = *(const uint32_t*)(ab1 + 16 + 2 * vv);
      }
    }
#pragma unroll
    for (int g = 0; g < 5; ++g) {
      // issue next fragment load (g+1 this kt, or g=0 of kt+1) before using bcur
      const int    gn   = (g < 4) ? g + 1 : 0;
      const size_t noff = (g < 4) ? BOFF(gn, kt) : BOFF(0, kt + 1);
      FragB16 bnxt;
      bnxt.u4[0] = *(const uint4*)(Wswz + noff);
      bnxt.u4[1] = *(const uint4*)(Wswz + noff + 8);

      acc[g * 2 + 0] = __builtin_amdgcn_wmma_f32_16x16x32_bf16(
          false, af0.v, false, bcur.v, (short)0, acc[g * 2 + 0], false, false);
      acc[g * 2 + 1] = __builtin_amdgcn_wmma_f32_16x16x32_bf16(
          false, af1.v, false, bcur.v, (short)0, acc[g * 2 + 1], false, false);

      bcur = bnxt;
    }
  }
  #undef BOFF

  __syncthreads();   // all waves done reading A_sh before h_sh overwrites it

  // ---- wave-local pointwise: gates, cell, hidden ----
#pragma unroll
  for (int rt = 0; rt < 2; ++rt) {
#pragma unroll
    for (int p = 0; p < 8; ++p) {
      const int row  = rt * 16 + p + ((lane >> 4) << 3);  // C/D layout: M = p (+8 hi lanes)
      const int j    = w * 16 + (lane & 15);              // N = lane%16 within tile
      const int r    = r0 + row;
      const int b    = r >> lvl;
      const int m    = r & (M - 1);
      const int node = s + m;

      float ig = acc[0 * 2 + rt][p] + b_i[j];
      float og = acc[1 * 2 + rt][p] + b_o[j];
      float ug = acc[2 * 2 + rt][p] + b_u[j];
      float f0 = acc[3 * 2 + rt][p] + b_f[j];
      float f1 = acc[4 * 2 + rt][p] + b_f[j];
      ig = 1.0f / (1.0f + __expf(-ig));
      og = 1.0f / (1.0f + __expf(-og));
      f0 = 1.0f / (1.0f + __expf(-f0));
      f1 = 1.0f / (1.0f + __expf(-f1));
      ug = tanhf(ug);

      float cL = 0.0f, cR = 0.0f;
      if (!leaf) {
        const int cl = e + 2 * m;
        cL = c_ws[((size_t)b * N_NODES + cl)     * HID + j];
        cR = c_ws[((size_t)b * N_NODES + cl + 1) * HID + j];
      }
      const float c = ig * ug + f0 * cL + f1 * cR;
      const float h = og * tanhf(c);

      const size_t base = ((size_t)b * N_NODES + node) * HID + j;
      c_ws[base] = c;
      h_ws[base] = f32_to_bf16(h);
      h_sh[row * H_STRIDE + j] = h;
    }
  }
  __syncthreads();

  // ---- fused classifier head: logits (5 x 256 dot) + softmax ----
  if (tid < ROWS_WG * NCLASS) {
    const int row = tid / NCLASS;
    const int cls = tid % NCLASS;
    float sum = b_s[cls];
    for (int j = 0; j < HID; ++j)
      sum += W_s[cls * HID + j] * h_sh[row * H_STRIDE + j];
    logit_sh[row * NCLASS + cls] = sum;
  }
  __syncthreads();
  if (tid < ROWS_WG) {
    const int row  = tid;
    const int r    = r0 + row;
    const int b    = r >> lvl;
    const int m    = r & (M - 1);
    const int node = s + m;
    float mx = logit_sh[row * NCLASS];
    for (int c = 1; c < NCLASS; ++c) mx = fmaxf(mx, logit_sh[row * NCLASS + c]);
    float ex[NCLASS], se = 0.0f;
    for (int c = 0; c < NCLASS; ++c) {
      ex[c] = __expf(logit_sh[row * NCLASS + c] - mx);
      se += ex[c];
    }
    const float inv = 1.0f / se;
    const size_t ob = ((size_t)b * N_NODES + node) * NCLASS;
    for (int c = 0; c < NCLASS; ++c) out[ob + c] = ex[c] * inv;
  }
}

// =====================================================================
extern "C" void kernel_launch(void* const* d_in, const int* in_sizes, int n_in,
                              void* d_out, int out_size, void* d_ws, size_t ws_size,
                              hipStream_t stream) {
  (void)in_sizes; (void)n_in; (void)out_size; (void)ws_size;
  const float* x   = (const float*)d_in[0];
  const float* W_i = (const float*)d_in[1];
  const float* U_i = (const float*)d_in[2];
  const float* b_i = (const float*)d_in[3];
  const float* W_f = (const float*)d_in[4];
  const float* U_f = (const float*)d_in[5];
  const float* b_f = (const float*)d_in[6];
  const float* W_o = (const float*)d_in[7];
  const float* U_o = (const float*)d_in[8];
  const float* b_o = (const float*)d_in[9];
  const float* W_u = (const float*)d_in[10];
  const float* U_u = (const float*)d_in[11];
  const float* b_u = (const float*)d_in[12];
  const float* W_s = (const float*)d_in[13];
  const float* b_s = (const float*)d_in[14];

  // workspace layout: swizzled weights (1.92 MB + pipeline pad) | h bf16 (134 MB) | c f32 (268 MB)
  char* ws = (char*)d_ws;
  unsigned short* Wswz = (unsigned short*)ws;
  size_t off = ((size_t)NT_TILES * KT_TILES * 32 * 16 * 2 + 4096 + 255) & ~(size_t)255;
  unsigned short* h_ws = (unsigned short*)(ws + off);
  off += (size_t)BATCH * N_NODES * HID * 2;
  off = (off + 255) & ~(size_t)255;
  float* c_ws = (float*)(ws + off);

  prep_weights<<<(NT_TILES * KT_TILES * 32) / 256, 256, 0, stream>>>(
      W_i, U_i, W_f, U_f, W_o, U_o, W_u, U_u, Wswz);

  for (int lvl = DEPTH - 1; lvl >= 0; --lvl) {
    const int M = 1 << lvl;
    const int R = BATCH * M;                 // rows at this level (multiple of 32... min 128)
    level_kernel<<<R / ROWS_WG, THREADS, 0, stream>>>(x, Wswz, b_i, b_f, b_o, b_u, W_s, b_s,
                                                      h_ws, c_ws, (float*)d_out, lvl);
  }
}
